// Qwen3VLVisualAdapter_50861002719465
// MI455X (gfx1250) — matrix operand order repaired and checked by reading, not compiled, not run on hardware
//
#include <hip/hip_runtime.h>
#include <math.h>

typedef __attribute__((ext_vector_type(16))) __bf16 v16bf;
typedef __attribute__((ext_vector_type(8)))  __bf16 v8bf;
typedef __attribute__((ext_vector_type(8)))  float  v8f;

#define D_DIM 4096
#define C_DIM 256
#define S_DIM 4096
#define EPS 1e-5f

// ---------------------------------------------------------------------------
// CDNA5 async global->LDS copy (ASYNCcnt-tracked, no VGPR round trip).
// the LDS destination offset is taken from the VDST VGPR; truncating a generic LDS pointer to 32 bits
// yields the LDS offset (aperture addressing, ISA 10.2).
// ---------------------------------------------------------------------------
__device__ __forceinline__ void async_ld_b128(unsigned lds_off, const void* g) {
    asm volatile("global_load_async_to_lds_b128 %0, %1, off"
                 :: "v"(lds_off), "v"(g) : "memory");
}
__device__ __forceinline__ void wait_async0() {
    asm volatile("s_wait_asynccnt 0x0" ::: "memory");
}

// ---------------------------------------------------------------------------
// K0: convert w_down [C,D] and w_up [D,C] to bf16 (layouts kept; both are
// K-contiguous per output column == WMMA B-operand order).
// ---------------------------------------------------------------------------
__global__ void prep_weights(const float* __restrict__ wd_f,
                             const float* __restrict__ wu_f,
                             __bf16* __restrict__ wd,
                             __bf16* __restrict__ wu, int n) {
    int i = blockIdx.x * blockDim.x + threadIdx.x;
    if (i < n) {
        wd[i] = (__bf16)wd_f[i];
        wu[i] = (__bf16)wu_f[i];
    }
}

// ---------------------------------------------------------------------------
// K1: per-token LayerNorm stats + sigmoid gate logit (fused single pass).
// ---------------------------------------------------------------------------
__global__ __launch_bounds__(256)
void ln_stats_gate(const float* __restrict__ x,
                   const float* __restrict__ wg,
                   const float* __restrict__ bg,
                   float* __restrict__ mean,
                   float* __restrict__ rstd,
                   float* __restrict__ gate) {
    const int t   = blockIdx.x;
    const int tid = threadIdx.x;
    const float4* row4 = (const float4*)(x + (size_t)t * D_DIM);
    const float4* wg4  = (const float4*)wg;

    float s = 0.f, ss = 0.f, g = 0.f;
    for (int i = tid; i < D_DIM / 4; i += 256) {
        float4 v = row4[i];
        float4 w = wg4[i];
        s  += v.x + v.y + v.z + v.w;
        ss += v.x * v.x + v.y * v.y + v.z * v.z + v.w * v.w;
        g  += v.x * w.x + v.y * w.y + v.z * w.z + v.w * w.w;
    }

    __shared__ float r0[256], r1[256], r2[256];
    r0[tid] = s; r1[tid] = ss; r2[tid] = g;
    __syncthreads();
    for (int st = 128; st > 0; st >>= 1) {
        if (tid < st) {
            r0[tid] += r0[tid + st];
            r1[tid] += r1[tid + st];
            r2[tid] += r2[tid + st];
        }
        __syncthreads();
    }
    if (tid == 0) {
        float mu  = r0[0] * (1.f / D_DIM);
        float var = r1[0] * (1.f / D_DIM) - mu * mu;
        mean[t] = mu;
        rstd[t] = rsqrtf(var + EPS);
        gate[t] = 1.f / (1.f + expf(-(r2[0] + bg[0])));
    }
}

// ---------------------------------------------------------------------------
// K2: down projection  h[t,c] = sum_d xn[t,d] * w_down[c,d]
// 256 thr = 8 waves; tile 64 tokens x 256 channels; K chunk 32.
// Double-buffered LDS (one barrier per K-step); B tile staged with
// async global->LDS; A normalized f32->bf16 through VGPRs.
// ---------------------------------------------------------------------------
__global__ __launch_bounds__(256)
void gemm_down(const float* __restrict__ x,
               const float* __restrict__ gamma,
               const float* __restrict__ beta,
               const __bf16* __restrict__ wd,
               const float* __restrict__ mean,
               const float* __restrict__ rstd,
               float* __restrict__ h) {
    __shared__ __attribute__((aligned(32))) __bf16 lA[2][64 * 32];
    __shared__ __attribute__((aligned(32))) __bf16 lB[2][C_DIM * 32];

    const int tid  = threadIdx.x;
    const int lane = tid & 31;
    const int wid  = tid >> 5;
    const int wm   = wid & 3;          // rows wm*16
    const int wn   = wid >> 2;         // cols wn*128
    const int tok0 = blockIdx.x * 64;

    // A staging coords: thread -> (row am, k-offset ak), 8 elems
    const int am = tid >> 2;
    const int ak = (tid & 3) * 8;
    const float mu = mean[tok0 + am];
    const float rs = rstd[tok0 + am];
    const float* aSrc = x + (size_t)(tok0 + am) * D_DIM + ak;
    const __bf16* bSrc = wd + (size_t)tid * D_DIM;   // column `tid`

    v8f acc[8] = {};

    // fragment addressing (wave32; A and B both use k = 16*(lane/16) + s)
    const int ar  = wm * 16 + (lane & 15);
    const int akb = (lane >> 4) * 16;
    const int bc  = lane & 15;
    const int bkb = (lane >> 4) * 16;

    auto stageA = [&](int k0, int p) {
        float4 a0 = *(const float4*)(aSrc + k0);
        float4 a1 = *(const float4*)(aSrc + k0 + 4);
        float4 gA = *(const float4*)(gamma + k0 + ak);
        float4 gB = *(const float4*)(gamma + k0 + ak + 4);
        float4 bA = *(const float4*)(beta  + k0 + ak);
        float4 bB = *(const float4*)(beta  + k0 + ak + 4);
        v8bf av;
        av[0] = (__bf16)((a0.x - mu) * rs * gA.x + bA.x);
        av[1] = (__bf16)((a0.y - mu) * rs * gA.y + bA.y);
        av[2] = (__bf16)((a0.z - mu) * rs * gA.z + bA.z);
        av[3] = (__bf16)((a0.w - mu) * rs * gA.w + bA.w);
        av[4] = (__bf16)((a1.x - mu) * rs * gB.x + bB.x);
        av[5] = (__bf16)((a1.y - mu) * rs * gB.y + bB.y);
        av[6] = (__bf16)((a1.z - mu) * rs * gB.z + bB.z);
        av[7] = (__bf16)((a1.w - mu) * rs * gB.w + bB.w);
        *(v8bf*)(&lA[p][am * 32 + ak]) = av;
        if (k0 + 32 < D_DIM)
            __builtin_prefetch((const void*)(aSrc + k0 + 32), 0, 1);
    };
    auto stageB = [&](int k0, int p) {
        unsigned dst = (unsigned)(size_t)(&lB[p][tid * 32]);
        const __bf16* src = bSrc + k0;
        async_ld_b128(dst,      src);
        async_ld_b128(dst + 16, src + 8);
        async_ld_b128(dst + 32, src + 16);
        async_ld_b128(dst + 48, src + 24);
    };

    stageA(0, 0);
    stageB(0, 0);

    int p = 0;
    for (int k0 = 0; k0 < D_DIM; k0 += 32) {
        wait_async0();          // our async stores to buf p have landed
        __syncthreads();        // everyone's staging of buf p done

        const int kn = k0 + 32;
        if (kn < D_DIM) {
            stageA(kn, p ^ 1);
            stageB(kn, p ^ 1);
        }

        union { v16bf v; v8bf q[2]; } afrag;
        afrag.q[0] = *(const v8bf*)(&lA[p][ar * 32 + akb]);
        afrag.q[1] = *(const v8bf*)(&lA[p][ar * 32 + akb + 8]);

#pragma unroll
        for (int g = 0; g < 2; g++) {
            v16bf bfrag[4];
#pragma unroll
            for (int t = 0; t < 4; t++)
                bfrag[t] = *(const v16bf*)(
                    &lB[p][(wn * 128 + (g * 4 + t) * 16 + bc) * 32 + bkb]);
#pragma unroll
            for (int t = 0; t < 4; t++)
                acc[g * 4 + t] = __builtin_amdgcn_wmma_f32_16x16x32_bf16(
                    false, afrag.v, false, bfrag[t], (short)0,
                    acc[g * 4 + t], false, false);
        }
        p ^= 1;
    }

#pragma unroll
    for (int t = 0; t < 8; t++) {
        int col = wn * 128 + t * 16 + (lane & 15);
#pragma unroll
        for (int v = 0; v < 8; v++) {
            int row = tok0 + wm * 16 + (lane >> 4) * 8 + v;
            h[(size_t)row * C_DIM + col] = acc[t][v];
        }
    }
}

// ---------------------------------------------------------------------------
// K3: depthwise conv1d (k=3, pad=1) along S + exact GELU -> bf16
// ---------------------------------------------------------------------------
__global__ __launch_bounds__(256)
void conv_gelu(const float* __restrict__ h,
               const float* __restrict__ wc,
               __bf16* __restrict__ g, int total) {
    int i = blockIdx.x * 256 + threadIdx.x;
    if (i >= total) return;
    int c = i & (C_DIM - 1);
    int t = i >> 8;
    int s = t & (S_DIM - 1);

    float w0 = wc[c * 3 + 0], w1 = wc[c * 3 + 1], w2 = wc[c * 3 + 2];
    float hm = (s > 0)         ? h[i - C_DIM] : 0.f;
    float h0 = h[i];
    float hp = (s < S_DIM - 1) ? h[i + C_DIM] : 0.f;
    float v  = hm * w0 + h0 * w1 + hp * w2;
    float ge = 0.5f * v * (1.f + erff(v * 0.70710678118654752f));
    g[i] = (__bf16)ge;
}

// ---------------------------------------------------------------------------
// K4: up projection + gated residual epilogue
//   out[t,d] = x[t,d] + gate[t] * sum_c g[t,c] * w_up[d,c]
// Both A and B tiles staged with async global->LDS; double-buffered.
// ---------------------------------------------------------------------------
__global__ __launch_bounds__(256)
void gemm_up(const __bf16* __restrict__ g,
             const __bf16* __restrict__ wu,
             const float* __restrict__ x,
             const float* __restrict__ gate,
             float* __restrict__ out) {
    __shared__ __attribute__((aligned(32))) __bf16 lA[2][64 * 32];
    __shared__ __attribute__((aligned(32))) __bf16 lB[2][256 * 32];

    const int tid  = threadIdx.x;
    const int lane = tid & 31;
    const int wid  = tid >> 5;
    const int wm   = wid & 3;
    const int wn   = wid >> 2;
    const int tok0 = blockIdx.x * 64;
    const int n0   = blockIdx.y * 256;

    const int am = tid >> 2;
    const int ak = (tid & 3) * 8;
    const __bf16* aSrc = g  + (size_t)(tok0 + am) * C_DIM + ak;
    const __bf16* bSrc = wu + (size_t)(n0 + tid) * C_DIM;

    v8f acc[8] = {};

    const int ar  = wm * 16 + (lane & 15);
    const int akb = (lane >> 4) * 16;
    const int bc  = lane & 15;
    const int bkb = (lane >> 4) * 16;

    auto stage = [&](int k0, int p) {
        async_ld_b128((unsigned)(size_t)(&lA[p][am * 32 + ak]), aSrc + k0);
        unsigned dst = (unsigned)(size_t)(&lB[p][tid * 32]);
        const __bf16* src = bSrc + k0;
        async_ld_b128(dst,      src);
        async_ld_b128(dst + 16, src + 8);
        async_ld_b128(dst + 32, src + 16);
        async_ld_b128(dst + 48, src + 24);
    };

    stage(0, 0);

    int p = 0;
    for (int k0 = 0; k0 < C_DIM; k0 += 32) {
        wait_async0();
        __syncthreads();

        const int kn = k0 + 32;
        if (kn < C_DIM) stage(kn, p ^ 1);

        union { v16bf v; v8bf q[2]; } afrag;
        afrag.q[0] = *(const v8bf*)(&lA[p][ar * 32 + akb]);
        afrag.q[1] = *(const v8bf*)(&lA[p][ar * 32 + akb + 8]);

#pragma unroll
        for (int gg = 0; gg < 2; gg++) {
            v16bf bfrag[4];
#pragma unroll
            for (int t = 0; t < 4; t++)
                bfrag[t] = *(const v16bf*)(
                    &lB[p][(wn * 128 + (gg * 4 + t) * 16 + bc) * 32 + bkb]);
#pragma unroll
            for (int t = 0; t < 4; t++)
                acc[gg * 4 + t] = __builtin_amdgcn_wmma_f32_16x16x32_bf16(
                    false, afrag.v, false, bfrag[t], (short)0,
                    acc[gg * 4 + t], false, false);
        }
        p ^= 1;
    }

#pragma unroll
    for (int t = 0; t < 8; t++) {
        int dcol = n0 + wn * 128 + t * 16 + (lane & 15);
#pragma unroll
        for (int v = 0; v < 8; v++) {
            int row = tok0 + wm * 16 + (lane >> 4) * 8 + v;
            size_t idx = (size_t)row * D_DIM + dcol;
            out[idx] = x[idx] + gate[row] * acc[t][v];
        }
    }
}

// ---------------------------------------------------------------------------
// Host launcher
// ---------------------------------------------------------------------------
extern "C" void kernel_launch(void* const* d_in, const int* in_sizes, int n_in,
                              void* d_out, int out_size, void* d_ws, size_t ws_size,
                              hipStream_t stream) {
    const float* x      = (const float*)d_in[0];
    const float* gamma  = (const float*)d_in[1];
    const float* beta   = (const float*)d_in[2];
    const float* w_down = (const float*)d_in[3];
    const float* w_conv = (const float*)d_in[4];
    const float* w_up   = (const float*)d_in[5];
    const float* w_gate = (const float*)d_in[6];
    const float* b_gate = (const float*)d_in[7];
    float* out = (float*)d_out;

    const int D = D_DIM, C = C_DIM;
    const int T = in_sizes[0] / D;    // B*S = 16384

    char* ws = (char*)d_ws;
    __bf16* wd = (__bf16*)ws;  ws += (size_t)C * D * sizeof(__bf16);
    __bf16* wu = (__bf16*)ws;  ws += (size_t)D * C * sizeof(__bf16);
    __bf16* gb = (__bf16*)ws;  ws += (size_t)T * C * sizeof(__bf16);
    float*  h  = (float*)ws;   ws += (size_t)T * C * sizeof(float);
    float*  mean = (float*)ws; ws += (size_t)T * sizeof(float);
    float*  rstd = (float*)ws; ws += (size_t)T * sizeof(float);
    float*  gate = (float*)ws; ws += (size_t)T * sizeof(float);

    const int NW = C * D;
    prep_weights<<<(NW + 255) / 256, 256, 0, stream>>>(w_down, w_up, wd, wu, NW);

    ln_stats_gate<<<T, 256, 0, stream>>>(x, w_gate, b_gate, mean, rstd, gate);

    gemm_down<<<T / 64, 256, 0, stream>>>(x, gamma, beta, wd, mean, rstd, h);

    conv_gelu<<<(T * C + 255) / 256, 256, 0, stream>>>(h, w_conv, gb, T * C);

    dim3 grid2(T / 64, D / 256);
    gemm_up<<<grid2, 256, 0, stream>>>(gb, wu, x, gate, out);
}
